// ResNet_13778255085764
// MI455X (gfx1250) — matrix operand order repaired and checked by reading, not compiled, not run on hardware
//
#include <hip/hip_runtime.h>
#include <hip/hip_bf16.h>

// ---------------------------------------------------------------------------
// Types for CDNA5 WMMA (wave32): D(16x16 f32) = A(16x32 bf16) x B(32x16 bf16) + C
// ---------------------------------------------------------------------------
typedef __attribute__((ext_vector_type(16))) __bf16 v16bf;
typedef __attribute__((ext_vector_type(8)))  float  v8f;

union Frag {          // 32 bytes, assembled from two 16B (b128) loads
    uint4 u[2];
    v16bf v;
};

// ---------------------------------------------------------------------------
// Weight pack: W[256][Cin][Ks] f32  ->  bf16 A-fragments.
// K flattened k-major: kk = k*Cin + ci, so a 32-chunk q = k*(Cin/32) + c0/32
// holds 32 consecutive channels at one tap.
// Contiguous A layout (16x32): lanes 0-15 hold k 0-15,
// lanes 16-31 hold k 16-31; element e of the per-lane v16bf maps to
// koff = half*16 + e.
// A index: (((mt*QT + q)*32 + lane)*16 + e)
// ---------------------------------------------------------------------------
__global__ __launch_bounds__(256) void pack_kernel(
    const float* __restrict__ W, __bf16* __restrict__ A, int Cin, int Ks, int total)
{
    int idx = blockIdx.x * 256 + threadIdx.x;
    if (idx >= total) return;
    int e    = idx & 15;
    int lane = (idx >> 4) & 31;
    int rest = idx >> 9;
    int QT   = (Cin * Ks) >> 5;
    int q    = rest % QT;
    int mt   = rest / QT;
    int h    = lane >> 4;
    int m    = mt * 16 + (lane & 15);
    int koff = (h << 4) + e;
    int CPK  = Cin >> 5;
    int k    = q / CPK;
    int ci   = (q - k * CPK) * 32 + koff;
    A[idx] = (__bf16)W[(m * Cin + ci) * Ks + k];
}

// ---------------------------------------------------------------------------
// Implicit-GEMM conv1d with WMMA.
//   X: [8][CIN][Lin] f32,  Y: [8][256][Lout] f32
//   grid = (Lout/NT, 8), block = 256 threads (8 waves)
//   wave w computes rows [32w, 32w+32) x NT positions (NT/16 n-tiles).
// Input tile staged in LDS as bf16, transposed [x][ci], row padded by 8
// elements so b128 reads across rows spread over all 64 banks.
// NT=64 for the dominant stride-1 convs amortizes A-fragment (weight)
// traffic: per 32-K chunk a wave issues 8 WMMAs vs 4 global + 8 DS b128s.
// ---------------------------------------------------------------------------
template <int CIN, int KSZ, int STRIDE, int NT, bool IN_RELU, bool RESID>
__global__ __launch_bounds__(256) void conv_wmma(
    const float* __restrict__ X, int Lin, int Lout,
    const __bf16* __restrict__ A, const float* __restrict__ bias,
    const float* __restrict__ resid, float* __restrict__ Y)
{
    constexpr int TILEW  = (NT - 1) * STRIDE + KSZ;
    constexpr int ROW    = CIN + 8;                 // bf16 elems; ROW*2 % 16 == 0
    constexpr int CHUNKS = KSZ * CIN / 32;
    constexpr int CPK    = CIN / 32;
    constexpr int PAD    = (KSZ == 3) ? 1 : 0;
    constexpr int NTILES = NT / 16;

    __shared__ __align__(16) __bf16 xs[TILEW * ROW];

    const int tid = threadIdx.x;
    const int b   = blockIdx.y;
    const int n0  = blockIdx.x * NT;

    // ---- stage input tile (f32 global -> bf16 LDS, transposed) ----
    for (int idx = tid; idx < TILEW * CIN; idx += 256) {
        int ci = idx / TILEW;
        int x  = idx - ci * TILEW;          // consecutive tid -> consecutive l
        int l  = n0 * STRIDE + x - PAD;
        float v = 0.0f;
        if (l >= 0 && l < Lin) v = X[(size_t)(b * CIN + ci) * Lin + l];
        if (IN_RELU) v = v > 0.0f ? v : 0.0f;
        xs[x * ROW + ci] = (__bf16)v;
    }
    __syncthreads();

    const int w    = tid >> 5;
    const int lane = tid & 31;
    const int col  = lane & 15;
    const int h    = lane >> 4;

    v8f acc[2][NTILES] = {};

    for (int q = 0; q < CHUNKS; ++q) {
        const int k  = q / CPK;
        const int c0 = (q - k * CPK) * 32;

        Frag a[2];
#pragma unroll
        for (int i = 0; i < 2; ++i) {
            size_t off = ((size_t)((2 * w + i) * CHUNKS + q) * 32 + lane) * 16;
            a[i].u[0] = *(const uint4*)(A + off);
            a[i].u[1] = *(const uint4*)(A + off + 8);
        }
        Frag bb[NTILES];
#pragma unroll
        for (int j = 0; j < NTILES; ++j) {
            const int xp = (j * 16 + col) * STRIDE + k;
            const __bf16* p = xs + xp * ROW + c0 + h * 16;   // 16B aligned
            bb[j].u[0] = *(const uint4*)p;
            bb[j].u[1] = *(const uint4*)(p + 8);
        }
#pragma unroll
        for (int i = 0; i < 2; ++i)
#pragma unroll
            for (int j = 0; j < NTILES; ++j)
                acc[i][j] = __builtin_amdgcn_wmma_f32_16x16x32_bf16(
                    false, a[i].v, false, bb[j].v, (short)0, acc[i][j], false, false);
    }

    // ---- epilogue: C/D layout: VGPR r -> row r + 8*half, col = lane&15 ----
#pragma unroll
    for (int i = 0; i < 2; ++i) {
        const int mt = 2 * w + i;
#pragma unroll
        for (int r = 0; r < 8; ++r) {
            const int m  = mt * 16 + r + 8 * h;
            const float bv = bias[m];
#pragma unroll
            for (int j = 0; j < NTILES; ++j) {
                const int l = n0 + j * 16 + col;
                float v = acc[i][j][r] + bv;
                if (RESID)   // nearest-x2 upsample residual (P4 = conv + repeat(P5))
                    v += resid[(size_t)(b * 256 + m) * (Lout >> 1) + (l >> 1)];
                Y[(size_t)(b * 256 + m) * Lout + l] = v;
            }
        }
    }
}

// ---------------------------------------------------------------------------
// In-place GroupNorm(32 groups, eps=1e-5) + ReLU.  grid=(32,8), block=256.
// ---------------------------------------------------------------------------
__global__ __launch_bounds__(256) void gnrelu_kernel(
    float* __restrict__ x, const float* __restrict__ gamma,
    const float* __restrict__ beta, int L)
{
    __shared__ float rs[256], rq[256];
    const int tid = threadIdx.x, g = blockIdx.x, b = blockIdx.y;

    float s = 0.0f, s2 = 0.0f;
    for (int c = 0; c < 8; ++c) {
        const float* p = x + (size_t)(b * 256 + g * 8 + c) * L;
        for (int l = tid; l < L; l += 256) { float v = p[l]; s += v; s2 += v * v; }
    }
    rs[tid] = s; rq[tid] = s2;
    __syncthreads();
    for (int off = 128; off > 0; off >>= 1) {
        if (tid < off) { rs[tid] += rs[tid + off]; rq[tid] += rq[tid + off]; }
        __syncthreads();
    }
    const float n    = 8.0f * (float)L;
    const float mean = rs[0] / n;
    float var = rq[0] / n - mean * mean;
    if (var < 0.0f) var = 0.0f;
    const float inv = rsqrtf(var + 1e-5f);

    for (int c = 0; c < 8; ++c) {
        const int ch = g * 8 + c;
        const float sc = gamma[ch] * inv;
        const float sh = beta[ch] - mean * sc;
        float* p = x + (size_t)(b * 256 + ch) * L;
        for (int l = tid; l < L; l += 256) {
            float v = p[l] * sc + sh;
            p[l] = v > 0.0f ? v : 0.0f;
        }
    }
}

// ---------------------------------------------------------------------------
// Output conv (Cout=6, K=3, pad=1) + optional sigmoid, written directly into
// d_out as [B, L*3, 2] slabs at row offset rowOff (anchor-folded transpose):
//   out[b, rowOff + l*3 + a, c] = act(o[b, a*2+c, l])
// ---------------------------------------------------------------------------
__global__ __launch_bounds__(256) void outconv_kernel(
    const float* __restrict__ x, const float* __restrict__ w,
    const float* __restrict__ bias, float* __restrict__ out,
    int L, int rowOff, int applySig)
{
    __shared__ float ws[6 * 256 * 3];
    const int tid = threadIdx.x, b = blockIdx.y;
    for (int i = tid; i < 4608; i += 256) ws[i] = w[i];
    __syncthreads();

    const int l = blockIdx.x * 256 + tid;   // L is a multiple of 256
    float acc[6];
#pragma unroll
    for (int co = 0; co < 6; ++co) acc[co] = bias[co];

    for (int ci = 0; ci < 256; ++ci) {
        const float* row = x + (size_t)(b * 256 + ci) * L;
        const float x0 = (l > 0)     ? row[l - 1] : 0.0f;
        const float x1 = row[l];
        const float x2 = (l < L - 1) ? row[l + 1] : 0.0f;
#pragma unroll
        for (int co = 0; co < 6; ++co) {
            const float* wr = &ws[(co * 256 + ci) * 3];
            acc[co] += wr[0] * x0 + wr[1] * x1 + wr[2] * x2;
        }
    }
    const int TOT = 47616;   // sum(L)*3 anchors
#pragma unroll
    for (int a = 0; a < 3; ++a)
#pragma unroll
        for (int c = 0; c < 2; ++c) {
            float v = acc[a * 2 + c];
            if (applySig) v = 1.0f / (1.0f + __expf(-v));
            out[((size_t)b * TOT + rowOff + l * 3 + a) * 2 + c] = v;
        }
}

// ---------------------------------------------------------------------------
// Host orchestration
// ---------------------------------------------------------------------------
extern "C" void kernel_launch(void* const* d_in, const int* in_sizes, int n_in,
                              void* d_out, int out_size, void* d_ws, size_t ws_size,
                              hipStream_t stream)
{
    (void)in_sizes; (void)n_in; (void)out_size; (void)ws_size;

    // inputs in setup_inputs()/make_params() insertion order
    const float* C4 = (const float*)d_in[0];  // [8,256,8192]
    const float* C5 = (const float*)d_in[1];  // [8,512,4096]
    const float *w51 = (const float*)d_in[2],  *b51 = (const float*)d_in[3];
    const float *w52 = (const float*)d_in[4],  *b52 = (const float*)d_in[5];
    const float *w41 = (const float*)d_in[6],  *b41 = (const float*)d_in[7];
    const float *w42 = (const float*)d_in[8],  *b42 = (const float*)d_in[9];
    const float *w6  = (const float*)d_in[10], *b6  = (const float*)d_in[11];
    const float *w7  = (const float*)d_in[12], *b7  = (const float*)d_in[13];
    const float *w8  = (const float*)d_in[14], *b8  = (const float*)d_in[15];
    const float *wc1 = (const float*)d_in[16], *bc1 = (const float*)d_in[17];
    const float *wc2 = (const float*)d_in[18], *bc2 = (const float*)d_in[19];
    const float *wco = (const float*)d_in[20], *bco = (const float*)d_in[21];
    const float *wr1 = (const float*)d_in[22], *br1 = (const float*)d_in[23];
    const float *wr2 = (const float*)d_in[24], *br2 = (const float*)d_in[25];
    const float *wro = (const float*)d_in[26], *bro = (const float*)d_in[27];
    const float *cg1 = (const float*)d_in[28], *cb1 = (const float*)d_in[29];
    const float *cg2 = (const float*)d_in[30], *cb2 = (const float*)d_in[31];
    const float *rg1 = (const float*)d_in[32], *rb1 = (const float*)d_in[33];
    const float *rg2 = (const float*)d_in[34], *rb2 = (const float*)d_in[35];

    // workspace layout (fp32 activations, then packed bf16 weights)
    const size_t S4 = (size_t)8 * 256 * 8192;   // 16,777,216
    const size_t S5 = S4 / 2, S6 = S4 / 4, S7 = S4 / 8, S8 = S4 / 16;
    float* ws  = (float*)d_ws;
    float* P5  = ws;            // [8,256,4096]
    float* P5o = P5  + S5;
    float* P4  = P5o + S5;      // reused as head H1 after P4_out
    float* P4o = P4  + S4;
    float* H2  = P4o + S4;
    float* P6  = H2  + S4;
    float* P7  = P6  + S6;
    float* P8  = P7  + S7;
    __bf16* pk = (__bf16*)(P8 + S8);

    const size_t K33 = 256 * 256 * 3;   // 196,608 elems
    __bf16* p51 = pk;                    // 256*512*1
    __bf16* p52 = p51 + 256 * 512;       // 256*256*3
    __bf16* p41 = p52 + K33;             // 256*256*1
    __bf16* p42 = p41 + 256 * 256;
    __bf16* p6w = p42 + K33;             // 256*512*3
    __bf16* p7w = p6w + 256 * 512 * 3;
    __bf16* p8w = p7w + K33;
    __bf16* pc1 = p8w + K33;
    __bf16* pc2 = pc1 + K33;
    __bf16* pr1 = pc2 + K33;
    __bf16* pr2 = pr1 + K33;

    auto pack = [&](const float* W, __bf16* A, int Cin, int Ks) {
        int total = 256 * Cin * Ks;
        pack_kernel<<<(total + 255) / 256, 256, 0, stream>>>(W, A, Cin, Ks, total);
    };
    pack(w51, p51, 512, 1);  pack(w52, p52, 256, 3);
    pack(w41, p41, 256, 1);  pack(w42, p42, 256, 3);
    pack(w6,  p6w, 512, 3);  pack(w7,  p7w, 256, 3);  pack(w8, p8w, 256, 3);
    pack(wc1, pc1, 256, 3);  pack(wc2, pc2, 256, 3);
    pack(wr1, pr1, 256, 3);  pack(wr2, pr2, 256, 3);

    // ---- FPN ----
    conv_wmma<512, 1, 1, 32, false, false>   // NT=32: LDS tile < 64KB for CIN=512
        <<<dim3(4096 / 32, 8), 256, 0, stream>>>(C5, 4096, 4096, p51, b51, nullptr, P5);
    conv_wmma<256, 1, 1, 64, false, true>
        <<<dim3(8192 / 64, 8), 256, 0, stream>>>(C4, 8192, 8192, p41, b41, P5, P4);
    conv_wmma<256, 3, 1, 64, false, false>
        <<<dim3(4096 / 64, 8), 256, 0, stream>>>(P5, 4096, 4096, p52, b52, nullptr, P5o);
    conv_wmma<256, 3, 1, 64, false, false>
        <<<dim3(8192 / 64, 8), 256, 0, stream>>>(P4, 8192, 8192, p42, b42, nullptr, P4o);
    conv_wmma<512, 3, 2, 16, false, false>   // NT=16 keeps LDS tile < 64KB
        <<<dim3(2048 / 16, 8), 256, 0, stream>>>(C5, 4096, 2048, p6w, b6, nullptr, P6);
    conv_wmma<256, 3, 2, 32, true, false>
        <<<dim3(1024 / 32, 8), 256, 0, stream>>>(P6, 2048, 1024, p7w, b7, nullptr, P7);
    conv_wmma<256, 3, 2, 32, true, false>
        <<<dim3(512 / 32, 8), 256, 0, stream>>>(P7, 1024, 512, p8w, b8, nullptr, P8);

    // ---- heads over 5 levels ----
    const float* fms[5] = {P4o, P5o, P6, P7, P8};
    const int    Ls[5]  = {8192, 4096, 2048, 1024, 512};
    const int    rOf[5] = {0, 24576, 36864, 43008, 46080};
    float* H1 = P4;   // P4 is dead after P4_out

    for (int hd = 0; hd < 2; ++hd) {
        const __bf16* c1p = hd ? pr1 : pc1;   const float* c1b = hd ? br1 : bc1;
        const __bf16* c2p = hd ? pr2 : pc2;   const float* c2b = hd ? br2 : bc2;
        const float*  g1  = hd ? rg1 : cg1;   const float* be1 = hd ? rb1 : cb1;
        const float*  g2  = hd ? rg2 : cg2;   const float* be2 = hd ? rb2 : cb2;
        const float*  ow  = hd ? wro : wco;   const float* ob  = hd ? bro : bco;
        float* outPtr = (float*)d_out + (hd ? (size_t)8 * 47616 * 2 : 0);

        for (int lv = 0; lv < 5; ++lv) {
            const int L = Ls[lv];   // 8192..512, all multiples of 64 and 256
            conv_wmma<256, 3, 1, 64, false, false>
                <<<dim3(L / 64, 8), 256, 0, stream>>>(fms[lv], L, L, c1p, c1b, nullptr, H1);
            gnrelu_kernel<<<dim3(32, 8), 256, 0, stream>>>(H1, g1, be1, L);
            conv_wmma<256, 3, 1, 64, false, false>
                <<<dim3(L / 64, 8), 256, 0, stream>>>(H1, L, L, c2p, c2b, nullptr, H2);
            gnrelu_kernel<<<dim3(32, 8), 256, 0, stream>>>(H2, g2, be2, L);
            outconv_kernel<<<dim3(L / 256, 8), 256, 0, stream>>>(
                H2, ow, ob, outPtr, L, rOf[lv], hd ? 0 : 1);
        }
    }
}
